// Actor_84069689852054
// MI455X (gfx1250) — matrix operand order repaired and checked by reading, not compiled, not run on hardware
//
#include <hip/hip_runtime.h>
#include <hip/hip_bf16.h>

typedef __attribute__((ext_vector_type(16))) _Float16     v16h;
typedef __attribute__((ext_vector_type(8)))  float        v8f;
typedef __attribute__((ext_vector_type(4)))  unsigned int v4u;
typedef __attribute__((ext_vector_type(4)))  int          v4i;
typedef __attribute__((ext_vector_type(8)))  int          v8i;

#define D_MODEL 128
#define D_INNER 512
#define D_UNIT  256
#define D_QUERY 360
#define N_NODES 1024
#define N_EDGES 16384
#define N_HEADS 16
#define D_K     16

#if defined(__has_builtin)
#  if __has_builtin(__builtin_amdgcn_tensor_load_to_lds)
#    define HAVE_TDM 1
#  endif
#endif

#ifdef __HIP_DEVICE_COMPILE__
#  ifdef HAVE_TDM
#    warning "gfx1250 device pass: __builtin_amdgcn_tensor_load_to_lds AVAILABLE (TDM path compiled)"
#  else
#    warning "gfx1250 device pass: __builtin_amdgcn_tensor_load_to_lds NOT available (manual LDS staging fallback)"
#  endif
#endif

// ---------------------------------------------------------------------------
// Tensor Data Mover: issue async 2-D tile (rows x cols of f16) global -> LDS.
// D# groups per CDNA5 ISA 8.3/8.4: count=1, type=2, data_size=2B,
// tile == viewed tensor, dim0 stride = row_stride. Completion via TENSORcnt.
// ---------------------------------------------------------------------------
__device__ __forceinline__ void tdm_issue_2d_f16(
    const _Float16* gaddr, unsigned lds_addr, unsigned rows, unsigned cols,
    unsigned row_stride_elems) {
#ifdef HAVE_TDM
    unsigned long long ga = (unsigned long long)(size_t)gaddr;
    union { unsigned int w[4]; v4u v; } g0;
    g0.w[0] = 1u;                                       // count=1, user desc
    g0.w[1] = lds_addr;                                 // lds_addr [63:32]
    g0.w[2] = (unsigned)(ga & 0xFFFFFFFFull);           // global_addr [95:64]
    g0.w[3] = (unsigned)((ga >> 32) & 0x1FFFFFFull)     // global_addr [120:96]
            | (2u << 30);                               // type=2 [127:126]
    unsigned long long td0 = cols, td1 = rows, st0 = row_stride_elems;
    union { unsigned long long q[4]; v8i v; } g1;
    g1.q[0] = (1ull << 16)                              // data_size=1 -> 2B
            | ((td0 & 0xFFFFull) << 48);                // tensor_dim0[15:0]
    g1.q[1] = ((td0 >> 16) & 0xFFFFull)                 // tensor_dim0[31:16]
            | ((td1 & 0xFFFFFFFFull) << 16)             // tensor_dim1 [111:80]
            | ((td0 & 0xFFFFull) << 48);                // tile_dim0 [127:112]
    g1.q[2] = (td1 & 0xFFFFull)                         // tile_dim1 [143:128]
            | ((st0 & 0xFFFFFFFFull) << 32);            // dim0_stride[31:0]
    g1.q[3] = ((st0 >> 32) & 0xFFFFull);                // dim0_stride[47:32]
    v4i z4 = {0, 0, 0, 0};
#  if __clang_major__ >= 23
    v8i z8 = {0, 0, 0, 0, 0, 0, 0, 0};
    __builtin_amdgcn_tensor_load_to_lds(g0.v, g1.v, z4, z4, z8, 0);
#  else
    __builtin_amdgcn_tensor_load_to_lds(g0.v, g1.v, z4, z4, 0);
#  endif
#else
    (void)gaddr; (void)lds_addr; (void)rows; (void)cols; (void)row_stride_elems;
#endif
}

// ---------------------------------------------------------------------------
// Small utility kernels
// ---------------------------------------------------------------------------
__global__ void fill_f32_kernel(float* p, float v, int n) {
    int i = blockIdx.x * blockDim.x + threadIdx.x;
    if (i < n) p[i] = v;
}

__global__ void f32_to_f16_kernel(const float* __restrict__ s, _Float16* __restrict__ d, int n) {
    int i = blockIdx.x * blockDim.x + threadIdx.x;
    if (i < n) d[i] = (_Float16)s[i];
}

// pack wq|wk|wv (each K x 256, f32) into one f16 K x 768 row-major matrix
__global__ void pack_qkv_kernel(const float* __restrict__ wq, const float* __restrict__ wk,
                                const float* __restrict__ wv, _Float16* __restrict__ dst,
                                int K) {
    int i = blockIdx.x * blockDim.x + threadIdx.x;
    if (i >= K * 768) return;
    int k = i / 768, c = i % 768;
    float v;
    if (c < 256)      v = wq[(size_t)k * 256 + c];
    else if (c < 512) v = wk[(size_t)k * 256 + (c - 256)];
    else              v = wv[(size_t)k * 256 + (c - 512)];
    dst[i] = (_Float16)v;
}

__global__ void deg_edges_kernel(const int* __restrict__ dst, float* deg, int e) {
    int i = blockIdx.x * blockDim.x + threadIdx.x;
    if (i < e) atomicAdd(&deg[dst[i]], 1.0f);
}

__global__ void rsqrt_deg_kernel(const float* __restrict__ deg, float* dinv, int n) {
    int i = blockIdx.x * blockDim.x + threadIdx.x;
    if (i < n) dinv[i] = rsqrtf(fmaxf(deg[i], 1.0f));
}

__global__ void matmul_plain_kernel(const float* __restrict__ A, const float* __restrict__ W,
                                    float* __restrict__ out, int N, int K, int F) {
    int i = blockIdx.x * blockDim.x + threadIdx.x;
    if (i >= N * F) return;
    int n = i / F, f = i % F;
    float s = 0.f;
    for (int k = 0; k < K; ++k) s += A[(size_t)n * K + k] * W[(size_t)k * F + f];
    out[i] = s;
}

__global__ void gcn_agg_kernel(const int* __restrict__ src, const int* __restrict__ dst,
                               const float* __restrict__ ht, const float* __restrict__ dinv,
                               float* out, int F) {
    int e = blockIdx.x;
    int f = threadIdx.x;
    int s = src[e], d = dst[e];
    float w = dinv[s] * dinv[d];
    atomicAdd(&out[(size_t)d * F + f], ht[(size_t)s * F + f] * w);
}

__global__ void gcn_self_kernel(const float* __restrict__ ht, const float* __restrict__ dinv,
                                const float* __restrict__ b, float* out, int F, int relu) {
    int n = blockIdx.x, f = threadIdx.x;
    float di = dinv[n];
    float v = out[(size_t)n * F + f] + ht[(size_t)n * F + f] * di * di + b[f];
    if (relu) v = fmaxf(v, 0.f);
    out[(size_t)n * F + f] = v;
}

// ---------------------------------------------------------------------------
// WMMA GEMM compute step: one A fragment (16x32 @ k0), four B tiles from an
// LDS 32x64 slab, 4 WMMA accumulations. A and B both in k = 16*(lane/16) + s order.
// ---------------------------------------------------------------------------
__device__ __forceinline__ void gemm_step(const _Float16* __restrict__ Bsl,
                                          const _Float16* __restrict__ arow,
                                          int k0, v8f* acc, int l, int lg) {
    v16h a;
#pragma unroll
    for (int j = 0; j < 8; ++j) {
        a[j]     = arow[k0 + lg * 16 + j];       // K = k0 + lg*16 + j
        a[8 + j] = arow[k0 + lg * 16 + 8 + j];   // K = k0 + lg*16 + 8 + j
    }
#pragma unroll
    for (int nt = 0; nt < 4; ++nt) {
        v16h b;
#pragma unroll
        for (int e = 0; e < 16; ++e)             // K row = lg*16 + e of slab
            b[e] = Bsl[(lg * 16 + e) * 64 + nt * 16 + l];
        acc[nt] = __builtin_amdgcn_wmma_f32_16x16x32_f16(false, a, false, b,
                                                         (short)0, acc[nt],
                                                         false, false);
    }
}

// C[M,N] = act(A[M,K] @ B[K,N] + bias); A,B f16 row-major. One wave per
// 16x64 C tile. B slabs (32x64) stream through a double-buffered LDS
// pipeline fed by the Tensor Data Mover (TENSORcnt-throttled); manual
// staging fallback when the builtin is absent. K%32==0, M%16==0, N%64==0.
template <bool RELU>
__global__ void __launch_bounds__(32) gemm_wmma_kernel(
    const _Float16* __restrict__ A, const _Float16* __restrict__ B,
    const float* __restrict__ bias, float* __restrict__ Cf, _Float16* __restrict__ Ch,
    int M, int N, int K) {
    int n0 = blockIdx.x * 64;
    int m0 = blockIdx.y * 16;
    int lane = threadIdx.x & 31;
    int l = lane & 15;
    int lg = lane >> 4;

    __shared__ _Float16 Bs[2][32 * 64];

    v8f z = {0.f, 0.f, 0.f, 0.f, 0.f, 0.f, 0.f, 0.f};
    v8f acc[4];
#pragma unroll
    for (int nt = 0; nt < 4; ++nt) acc[nt] = z;

    const _Float16* arow = A + (size_t)(m0 + l) * K;

#ifdef HAVE_TDM
    // prologue: DMA first slab, then pipeline issue(i+1) / wait(i) / compute(i)
    tdm_issue_2d_f16(B + n0, (unsigned)(size_t)&Bs[0][0], 32u, 64u, (unsigned)N);
    int it = 0;
    for (int k0 = 0; k0 < K; k0 += 32, ++it) {
        int cur = it & 1;
        __builtin_prefetch(arow + k0 + 64, 0, 1);
        if (k0 + 32 < K) {
            tdm_issue_2d_f16(B + (size_t)(k0 + 32) * N + n0,
                             (unsigned)(size_t)&Bs[cur ^ 1][0], 32u, 64u, (unsigned)N);
            __builtin_amdgcn_s_wait_tensorcnt(1);  // slab "cur" resident; next in flight
        } else {
            __builtin_amdgcn_s_wait_tensorcnt(0);
        }
        gemm_step(&Bs[cur][0], arow, k0, acc, l, lg);
    }
#else
    for (int k0 = 0; k0 < K; k0 += 32) {
        __syncthreads();
        for (int i = lane; i < 32 * 64; i += 32)
            Bs[0][i] = B[(size_t)(k0 + (i >> 6)) * N + n0 + (i & 63)];
        __syncthreads();
        __builtin_prefetch(arow + k0 + 64, 0, 1);
        gemm_step(&Bs[0][0], arow, k0, acc, l, lg);
    }
#endif

    int rb = m0 + lg * 8;
#pragma unroll
    for (int nt = 0; nt < 4; ++nt) {
        int n = n0 + nt * 16 + l;
        float bv = bias ? bias[n] : 0.f;
#pragma unroll
        for (int v = 0; v < 8; ++v) {
            float val = acc[nt][v] + bv;
            if (RELU) val = fmaxf(val, 0.f);
            if (Cf) Cf[(size_t)(rb + v) * N + n] = val;
            if (Ch) Ch[(size_t)(rb + v) * N + n] = (_Float16)val;
        }
    }
}

// ---------------------------------------------------------------------------
// Flash attention over packed QKV [N, 768] f16 (q=cols 0..255, k=256..511,
// v=512..767). One wave per (16-query tile, head); d_k=16 padded to K=32;
// 32 keys/chunk; online softmax with 16-lane shfl_xor; P@V via WMMA.
// ---------------------------------------------------------------------------
__global__ void __launch_bounds__(32) attn_wmma_kernel(
    const _Float16* __restrict__ qkv, int ld, float* __restrict__ out, int N) {
    int q0 = blockIdx.x * 16;
    int head = blockIdx.y;
    int hoff = head * D_K;
    int lane = threadIdx.x & 31;
    int l = lane & 15;
    int lg = lane >> 4;

    __shared__ _Float16 P[16][32];

    v16h aq;
#pragma unroll
    for (int j = 0; j < 8; ++j) {
        int d = lg * 8 + j;
        aq[j]     = (_Float16)(0.25f * (float)qkv[(size_t)(q0 + l) * ld + hoff + d]);
        aq[8 + j] = (_Float16)0.f;
    }

    v8f o = {0.f, 0.f, 0.f, 0.f, 0.f, 0.f, 0.f, 0.f};
    float mrow[8], lrow[8];
#pragma unroll
    for (int v = 0; v < 8; ++v) { mrow[v] = -3.0e38f; lrow[v] = 0.f; }

    for (int kb = 0; kb < N; kb += 32) {
        v16h bk0, bk1;
#pragma unroll
        for (int e = 0; e < 16; ++e) {
            if (e < 8) {
                bk0[e] = qkv[(size_t)(kb + l) * ld + 256 + hoff + lg * 8 + e];
                bk1[e] = qkv[(size_t)(kb + 16 + l) * ld + 256 + hoff + lg * 8 + e];
            } else {
                bk0[e] = (_Float16)0.f;
                bk1[e] = (_Float16)0.f;
            }
        }
        v8f z = {0.f, 0.f, 0.f, 0.f, 0.f, 0.f, 0.f, 0.f};
        v8f s0 = __builtin_amdgcn_wmma_f32_16x16x32_f16(false, aq, false, bk0,
                                                        (short)0, z, false, false);
        v8f s1 = __builtin_amdgcn_wmma_f32_16x16x32_f16(false, aq, false, bk1,
                                                        (short)0, z, false, false);
#pragma unroll
        for (int v = 0; v < 8; ++v) {
            float x0 = s0[v], x1 = s1[v];
            float mx = fmaxf(x0, x1);
            for (int m = 8; m >= 1; m >>= 1) mx = fmaxf(mx, __shfl_xor(mx, m, 16));
            float mnew = fmaxf(mrow[v], mx);
            float p0 = __expf(x0 - mnew);
            float p1 = __expf(x1 - mnew);
            float rs = p0 + p1;
            for (int m = 8; m >= 1; m >>= 1) rs += __shfl_xor(rs, m, 16);
            float sc = __expf(mrow[v] - mnew);
            lrow[v] = lrow[v] * sc + rs;
            mrow[v] = mnew;
            o[v] *= sc;
            P[lg * 8 + v][l]      = (_Float16)p0;
            P[lg * 8 + v][16 + l] = (_Float16)p1;
        }
        __syncthreads();
        v16h ap, bv;
#pragma unroll
        for (int j = 0; j < 8; ++j) {
            ap[j]     = P[l][lg * 16 + j];
            ap[8 + j] = P[l][lg * 16 + 8 + j];
        }
#pragma unroll
        for (int e = 0; e < 16; ++e)
            bv[e] = qkv[(size_t)(kb + lg * 16 + e) * ld + 512 + hoff + l];
        o = __builtin_amdgcn_wmma_f32_16x16x32_f16(false, ap, false, bv,
                                                   (short)0, o, false, false);
        __syncthreads();
    }

#pragma unroll
    for (int v = 0; v < 8; ++v)
        out[(size_t)(q0 + lg * 8 + v) * 256 + hoff + l] = o[v] / lrow[v];
}

// ---------------------------------------------------------------------------
// Post-LN: h = LN(h + delta) * g + b, also emits f16 copy. 128 thr / row.
// ---------------------------------------------------------------------------
__global__ void __launch_bounds__(128) ln_residual_kernel(
    float* __restrict__ h, const float* __restrict__ delta,
    const float* __restrict__ g, const float* __restrict__ b,
    _Float16* __restrict__ h16) {
    int row = blockIdx.x;
    int t = threadIdx.x;
    __shared__ float red[128];
    float x = h[(size_t)row * D_MODEL + t] + delta[(size_t)row * D_MODEL + t];
    red[t] = x;
    __syncthreads();
    for (int s = 64; s > 0; s >>= 1) {
        if (t < s) red[t] += red[t + s];
        __syncthreads();
    }
    float mean = red[0] / (float)D_MODEL;
    __syncthreads();
    float d = x - mean;
    red[t] = d * d;
    __syncthreads();
    for (int s = 64; s > 0; s >>= 1) {
        if (t < s) red[t] += red[t + s];
        __syncthreads();
    }
    float var = red[0] / (float)D_MODEL;
    float y = d * rsqrtf(var + 1e-6f) * g[t] + b[t];
    h[(size_t)row * D_MODEL + t] = y;
    h16[(size_t)row * D_MODEL + t] = (_Float16)y;
}

// ---------------------------------------------------------------------------
// Serial pointer decoder: one 256-thread block, 1024 dependent steps.
// ---------------------------------------------------------------------------
__device__ __forceinline__ float hash_u01(unsigned x) {
    x ^= x >> 17; x *= 0xed5ad4bbu; x ^= x >> 11; x *= 0xac4c1b51u;
    x ^= x >> 15; x *= 0x31848babu; x ^= x >> 14;
    return ((float)(x >> 8) + 0.5f) * (1.0f / 16777216.0f);
}

__device__ void compute_logits_dev(const float* __restrict__ enc_r,
                                   const float* qproj, const float* vvec,
                                   const unsigned* visited, float* logits, int t) {
    for (int n = t; n < N_NODES; n += 256) {
        const float* er = enc_r + (size_t)n * D_UNIT;
        float s = 0.f;
        for (int u = 0; u < D_UNIT; ++u) s += tanhf(er[u] + qproj[u]) * vvec[u];
        logits[n] = ((visited[n >> 5] >> (n & 31)) & 1u) ? -1e9f : s;
    }
}

__device__ int sample_step_dev(float* logits, float* red, int* redi, int step,
                               int t, float* lpterm) {
    float best = -3.0e38f;
    int bi = 0;
    for (int n = t; n < N_NODES; n += 256) {
        unsigned hseed = ((unsigned)(step * N_NODES + n)) * 2654435761u + 42u;
        float u = hash_u01(hseed);
        float gnoise = -logf(-logf(u));
        float v = logits[n] + gnoise;
        if (v > best) { best = v; bi = n; }
    }
    red[t] = best; redi[t] = bi;
    __syncthreads();
    for (int s = 128; s > 0; s >>= 1) {
        if (t < s && red[t + s] > red[t]) { red[t] = red[t + s]; redi[t] = redi[t + s]; }
        __syncthreads();
    }
    int idx = redi[0];
    __syncthreads();
    float mx = -3.0e38f;
    for (int n = t; n < N_NODES; n += 256) mx = fmaxf(mx, logits[n]);
    red[t] = mx;
    __syncthreads();
    for (int s = 128; s > 0; s >>= 1) {
        if (t < s) red[t] = fmaxf(red[t], red[t + s]);
        __syncthreads();
    }
    float M = red[0];
    __syncthreads();
    float se = 0.f;
    for (int n = t; n < N_NODES; n += 256) se += __expf(logits[n] - M);
    red[t] = se;
    __syncthreads();
    for (int s = 128; s > 0; s >>= 1) {
        if (t < s) red[t] += red[t + s];
        __syncthreads();
    }
    float S = red[0];
    __syncthreads();
    *lpterm = logits[idx] - (M + logf(S));
    return idx;
}

__global__ void __launch_bounds__(256) decode_kernel(
    const float* __restrict__ enc, const float* __restrict__ enc_r,
    const float* __restrict__ q_l1, const float* __restrict__ ctxW,
    const float* __restrict__ ptr_wq, const float* __restrict__ ptr_bq,
    const float* __restrict__ ptr_v, const float* __restrict__ start_bq,
    const float* __restrict__ start_v, float* __restrict__ out) {
    const int T = 256;
    int t = threadIdx.x;
    __shared__ float logits[N_NODES];
    __shared__ float red[256];
    __shared__ int redi[256];
    __shared__ float context[D_QUERY], fq[D_QUERY], resid[D_QUERY];
    __shared__ float qproj[D_UNIT], q1[D_MODEL], vsh[D_UNIT];
    __shared__ unsigned visited[32];
    __shared__ float lp_sh;
    __shared__ int cur_sh;

    for (int j = t; j < D_QUERY; j += T) context[j] = 0.f;
    if (t < 32) visited[t] = 0u;
    if (t == 0) lp_sh = 0.f;
    for (int j = t; j < D_UNIT; j += T) { qproj[j] = start_bq[j]; vsh[j] = start_v[j]; }
    __syncthreads();

    compute_logits_dev(enc_r, qproj, vsh, visited, logits, t);
    __syncthreads();
    float lpterm;
    int idx = sample_step_dev(logits, red, redi, 0, t, &lpterm);
    if (t == 0) {
        lp_sh += lpterm;
        visited[idx >> 5] |= (1u << (idx & 31));
        out[0] = (float)idx;
        cur_sh = idx;
    }
    __syncthreads();
    for (int j = t; j < D_MODEL; j += T) q1[j] = enc[(size_t)cur_sh * D_MODEL + j];
    for (int j = t; j < D_UNIT; j += T) vsh[j] = ptr_v[j];
    __syncthreads();

    for (int step = 1; step < N_NODES; ++step) {
        for (int j = t; j < D_QUERY; j += T) {
            float s = 0.f;
            for (int k = 0; k < D_MODEL; ++k) s += q1[k] * q_l1[(size_t)k * D_QUERY + j];
            resid[j] = s;
        }
        __syncthreads();
        for (int j = t; j < D_QUERY; j += T) {
            float s = 0.f;
            for (int k = 0; k < D_QUERY; ++k) {
                float r = resid[k];
                r = r > 0.f ? r : 0.f;
                s += r * ctxW[(size_t)k * D_QUERY + j];
            }
            float c = fmaxf(context[j], s);
            context[j] = c;
            fq[j] = fmaxf(resid[j] + c, 0.f);
        }
        __syncthreads();
        for (int j = t; j < D_UNIT; j += T) {
            float s = ptr_bq[j];
            for (int k = 0; k < D_QUERY; ++k) s += fq[k] * ptr_wq[(size_t)k * D_UNIT + j];
            qproj[j] = s;
        }
        __syncthreads();
        compute_logits_dev(enc_r, qproj, vsh, visited, logits, t);
        __syncthreads();
        idx = sample_step_dev(logits, red, redi, step, t, &lpterm);
        if (t == 0) {
            lp_sh += lpterm;
            visited[idx >> 5] |= (1u << (idx & 31));
            out[step] = (float)idx;
            cur_sh = idx;
        }
        __syncthreads();
        for (int j = t; j < D_MODEL; j += T) q1[j] = enc[(size_t)cur_sh * D_MODEL + j];
        __syncthreads();
    }
    if (t == 0) out[N_NODES] = lp_sh;
}

// ---------------------------------------------------------------------------
// Host launch
// ---------------------------------------------------------------------------
static inline int ceil_div(int a, int b) { return (a + b - 1) / b; }

extern "C" void kernel_launch(void* const* d_in, const int* in_sizes, int n_in,
                              void* d_out, int out_size, void* d_ws, size_t ws_size,
                              hipStream_t stream) {
    // jax-tree (sorted key) flatten order of setup_inputs():
    const int*   edge   = (const int*)d_in[0];
    const float* br     = (const float*)d_in[1];
    const float* ctxW   = (const float*)d_in[2];
    struct EncP { const float *b_ff1,*b_ff2,*ln1_b,*ln1_g,*ln2_b,*ln2_g,
                              *w_ff1,*w_ff2,*wk,*wo,*wq,*wv; } enc[3];
    for (int l = 0; l < 3; ++l) {
        int b = 3 + l * 12;
        enc[l].b_ff1 = (const float*)d_in[b + 0];
        enc[l].b_ff2 = (const float*)d_in[b + 1];
        enc[l].ln1_b = (const float*)d_in[b + 2];
        enc[l].ln1_g = (const float*)d_in[b + 3];
        enc[l].ln2_b = (const float*)d_in[b + 4];
        enc[l].ln2_g = (const float*)d_in[b + 5];
        enc[l].w_ff1 = (const float*)d_in[b + 6];
        enc[l].w_ff2 = (const float*)d_in[b + 7];
        enc[l].wk    = (const float*)d_in[b + 8];
        enc[l].wo    = (const float*)d_in[b + 9];
        enc[l].wq    = (const float*)d_in[b + 10];
        enc[l].wv    = (const float*)d_in[b + 11];
    }
    const float* gcn_b1   = (const float*)d_in[39];
    const float* gcn_b2   = (const float*)d_in[40];
    const float* gcn_w1   = (const float*)d_in[41];
    const float* gcn_w2   = (const float*)d_in[42];
    const float* ptr_bq   = (const float*)d_in[43];
    const float* ptr_v    = (const float*)d_in[44];
    const float* ptr_wq   = (const float*)d_in[45];
    const float* q_l1     = (const float*)d_in[46];
    const float* start_bq = (const float*)d_in[47];
    const float* start_v  = (const float*)d_in[48];
    const float* wr       = (const float*)d_in[50];
    const float* x        = (const float*)d_in[51];
    const int* srcE = edge;
    const int* dstE = edge + N_EDGES;

    char* wp = (char*)d_ws;
    auto alloc = [&](size_t bytes) -> void* {
        void* p = (void*)wp;
        wp += (bytes + 255) & ~(size_t)255;
        return p;
    };
    float* deg    = (float*)alloc(N_NODES * 4);
    float* dinv   = (float*)alloc(N_NODES * 4);
    float* ht     = (float*)alloc((size_t)N_NODES * D_MODEL * 4);
    float* h      = (float*)alloc((size_t)N_NODES * D_MODEL * 4);
    float* gtmp   = (float*)alloc((size_t)N_NODES * D_INNER * 4);
    float* enc_r  = (float*)alloc((size_t)N_NODES * D_UNIT * 4);
    float* attn_o = (float*)alloc((size_t)N_NODES * 256 * 4);
    _Float16* h16    = (_Float16*)alloc((size_t)N_NODES * D_MODEL * 2);
    _Float16* qkv16  = (_Float16*)alloc((size_t)N_NODES * 768 * 2);
    _Float16* o16    = (_Float16*)alloc((size_t)N_NODES * 256 * 2);
    _Float16* f16buf = (_Float16*)alloc((size_t)N_NODES * D_INNER * 2);
    _Float16* w16    = (_Float16*)alloc((size_t)(128 * 768) * 2);
    (void)ws_size; (void)in_sizes; (void)n_in; (void)out_size;

    auto cvt = [&](const float* s, _Float16* d, int n) {
        f32_to_f16_kernel<<<ceil_div(n, 256), 256, 0, stream>>>(s, d, n);
    };
    auto gemm = [&](const _Float16* A, const _Float16* B, const float* bias,
                    float* Cf, _Float16* Ch, int M, int N, int K, bool relu) {
        dim3 grid(N / 64, M / 16);
        if (relu)
            gemm_wmma_kernel<true><<<grid, 32, 0, stream>>>(A, B, bias, Cf, Ch, M, N, K);
        else
            gemm_wmma_kernel<false><<<grid, 32, 0, stream>>>(A, B, bias, Cf, Ch, M, N, K);
    };

    // ---- GCN ----
    fill_f32_kernel<<<ceil_div(N_NODES, 256), 256, 0, stream>>>(deg, 1.0f, N_NODES);
    deg_edges_kernel<<<ceil_div(N_EDGES, 256), 256, 0, stream>>>(dstE, deg, N_EDGES);
    rsqrt_deg_kernel<<<ceil_div(N_NODES, 256), 256, 0, stream>>>(deg, dinv, N_NODES);
    matmul_plain_kernel<<<ceil_div(N_NODES * 64, 256), 256, 0, stream>>>(x, gcn_w1, ht, N_NODES, 10, 64);
    fill_f32_kernel<<<ceil_div(N_NODES * 64, 256), 256, 0, stream>>>(gtmp, 0.f, N_NODES * 64);
    gcn_agg_kernel<<<N_EDGES, 64, 0, stream>>>(srcE, dstE, ht, dinv, gtmp, 64);
    gcn_self_kernel<<<N_NODES, 64, 0, stream>>>(ht, dinv, gcn_b1, gtmp, 64, 1);
    matmul_plain_kernel<<<ceil_div(N_NODES * 128, 256), 256, 0, stream>>>(gtmp, gcn_w2, ht, N_NODES, 64, 128);
    fill_f32_kernel<<<ceil_div(N_NODES * 128, 256), 256, 0, stream>>>(h, 0.f, N_NODES * 128);
    gcn_agg_kernel<<<N_EDGES, 128, 0, stream>>>(srcE, dstE, ht, dinv, h, 128);
    gcn_self_kernel<<<N_NODES, 128, 0, stream>>>(ht, dinv, gcn_b2, h, 128, 0);
    cvt(h, h16, N_NODES * D_MODEL);

    // ---- transformer encoder ----
    for (int l = 0; l < 3; ++l) {
        pack_qkv_kernel<<<ceil_div(D_MODEL * 768, 256), 256, 0, stream>>>(
            enc[l].wq, enc[l].wk, enc[l].wv, w16, D_MODEL);
        gemm(h16, w16, nullptr, nullptr, qkv16, N_NODES, 768, D_MODEL, false);

        attn_wmma_kernel<<<dim3(N_NODES / 16, N_HEADS), 32, 0, stream>>>(qkv16, 768, attn_o, N_NODES);

        cvt(attn_o, o16, N_NODES * 256);
        cvt(enc[l].wo, w16, 256 * D_MODEL);
        gemm(o16, w16, nullptr, gtmp, nullptr, N_NODES, D_MODEL, 256, false);
        ln_residual_kernel<<<N_NODES, D_MODEL, 0, stream>>>(h, gtmp, enc[l].ln1_g, enc[l].ln1_b, h16);

        cvt(enc[l].w_ff1, w16, D_MODEL * D_INNER);
        gemm(h16, w16, enc[l].b_ff1, nullptr, f16buf, N_NODES, D_INNER, D_MODEL, true);
        cvt(enc[l].w_ff2, w16, D_INNER * D_MODEL);
        gemm(f16buf, w16, enc[l].b_ff2, gtmp, nullptr, N_NODES, D_MODEL, D_INNER, false);
        ln_residual_kernel<<<N_NODES, D_MODEL, 0, stream>>>(h, gtmp, enc[l].ln2_g, enc[l].ln2_b, h16);
    }

    cvt(wr, w16, D_MODEL * D_UNIT);
    gemm(h16, w16, br, enc_r, nullptr, N_NODES, D_UNIT, D_MODEL, false);

    decode_kernel<<<1, 256, 0, stream>>>(h, enc_r, q_l1, ctxW, ptr_wq, ptr_bq,
                                         ptr_v, start_bq, start_v, (float*)d_out);
}
